// SelfAttentionModule_19207093748298
// MI455X (gfx1250) — compile-verified, matrix operand order checked by reading, not run on hardware
//
#include <hip/hip_runtime.h>

#define B_  4
#define C_  256
#define N_  2048
#define H_  8
#define D_  32
#define EPS_ 1e-5f

typedef __attribute__((ext_vector_type(16))) __bf16 v16bf;
typedef __attribute__((ext_vector_type(8)))  float  v8f;
typedef unsigned short u16w;
typedef unsigned int   u32w;

// ---------- helpers ----------

__device__ __forceinline__ u16w f2bf(float f) {
    u32w u = __float_as_uint(f);
    u += 0x7FFFu + ((u >> 16) & 1u);       // round to nearest even
    return (u16w)(u >> 16);
}

union FragU { u16w us[16]; v16bf v; uint4 q[2]; };

// Load a 16x32 bf16 A-fragment (or the mirrored 32x16 B-fragment) from a
// row-major [row][>=32] bf16 buffer. Per the CDNA5 16-bit A layout:
//   lanes 0-15 : row = lane,    K elems {0..7, 16..23}
//   lanes 16-31: row = lane-16, K elems {8..15, 24..31}
// Each lane does two contiguous 16-byte loads.
__device__ __forceinline__ v16bf load_frag(const u16w* __restrict__ base,
                                           int stride_elems, int lane) {
    const int row = lane & 15;
    const int kb  = (lane >> 4) << 3;      // 0 or 8
    const u16w* p = base + row * stride_elems + kb;
    FragU f;
    f.q[0] = *(const uint4*)(p);
    f.q[1] = *(const uint4*)(p + 16);
    return f.v;
}

__device__ __forceinline__ v8f wmma_bf16(v16bf a, v16bf b, v8f c) {
    return __builtin_amdgcn_wmma_f32_16x16x32_bf16(
        false, a, false, b, (short)0, c, false, false);
}

// Async copy of 16 bytes per lane from global to LDS (GVS addressing).
// Tracked on ASYNCcnt; completes out-of-band of LOADcnt.
__device__ __forceinline__ void async_copy16(u16w* lds_base, u32w lds_off,
                                             const u16w* gbase, u32w g_off) {
    u32w lds_addr = (u32w)(uintptr_t)lds_base + lds_off;  // low 32 bits = LDS offset
    asm volatile("global_load_async_to_lds_b128 %0, %1, %2"
                 :: "v"(lds_addr), "v"(g_off), "s"(gbase)
                 : "memory");
}

__device__ __forceinline__ void wait_asynccnt0() {
    asm volatile("s_wait_asynccnt 0x0" ::: "memory");
}

// ---------- kernel 1: fp32 -> bf16 conversions + x transpose ----------

__global__ void __launch_bounds__(256)
prep_kernel(const float* __restrict__ x, const float* __restrict__ wqkv,
            const float* __restrict__ wproj,
            u16w* __restrict__ xT, u16w* __restrict__ wq16, u16w* __restrict__ wp16) {
    int i = blockIdx.x * 256 + threadIdx.x;
    // x [B,C,N] -> xT [B,N,C] bf16
    if (i < B_ * C_ * N_) {
        int n = i % N_;
        int c = (i / N_) % C_;
        int b = i / (N_ * C_);
        xT[(b * N_ + n) * C_ + c] = f2bf(x[i]);
    }
    if (i < 3 * C_ * C_) wq16[i] = f2bf(wqkv[i]);
    if (i < C_ * C_)     wp16[i] = f2bf(wproj[i]);
}

// ---------- kernel 2: QKV GEMM  qkv[o,n] = sum_c W[o,c] * x[b,c,n] + bias ----------
// One wave computes a 16(o) x 64(n) tile: 8 K-steps x 4 n-tiles = 32 WMMAs.

__global__ void __launch_bounds__(256)
qkv_gemm_kernel(const u16w* __restrict__ wq, const float* __restrict__ bq,
                const u16w* __restrict__ xT,
                u16w* __restrict__ qb, u16w* __restrict__ kbuf, u16w* __restrict__ vT) {
    const int lane = threadIdx.x & 31;
    const int wg   = blockIdx.x * 8 + (threadIdx.x >> 5);
    const int b    = wg / (48 * 32);
    const int rem  = wg % (48 * 32);
    const int ot   = rem >> 5;             // 48 o-tiles of 16 (over 768 rows)
    const int ng   = rem & 31;             // 32 n-groups of 64
    const int obase = ot << 4;
    const int nbase = ng << 6;

    v8f zero = {};
    v8f acc[4] = {zero, zero, zero, zero};

    const u16w* abase = wq + obase * C_;
    for (int kc = 0; kc < C_; kc += 32) {
        if (kc + 32 < C_) __builtin_prefetch(abase + kc + 32, 0, 3);
        v16bf a = load_frag(abase + kc, C_, lane);
#pragma unroll
        for (int j = 0; j < 4; ++j) {
            const u16w* bb = xT + (size_t)(b * N_ + nbase + (j << 4)) * C_ + kc;
            v16bf bf = load_frag(bb, C_, lane);
            acc[j] = wmma_bf16(a, bf, acc[j]);
        }
    }

    const int hi   = lane >> 4;
    const int ncol = lane & 15;
    const int which = obase >> 8;          // 0=q 1=k 2=v (tiles never straddle)
#pragma unroll
    for (int r = 0; r < 8; ++r) {
        int o  = obase + r + (hi << 3);
        int c  = o & 255;
        int hh = c >> 5, dd = c & 31;
        float bias = bq[o];
#pragma unroll
        for (int j = 0; j < 4; ++j) {
            int n = nbase + (j << 4) + ncol;
            u16w val = f2bf(acc[j][r] + bias);
            if (which == 0)      qb  [((size_t)(b * H_ + hh) * N_ + n) * D_ + dd] = val;
            else if (which == 1) kbuf[((size_t)(b * H_ + hh) * N_ + n) * D_ + dd] = val;
            else                 vT  [((size_t)(b * H_ + hh) * D_ + dd) * N_ + n] = val;
        }
    }
}

// ---------- kernel 3: flash attention ----------
// 8 waves per block share one (b,h); K/V 32-key tiles are streamed into LDS
// with double-buffered GLOBAL_LOAD_ASYNC_TO_LDS_B128 (waves 0/1 issue, all
// waves consume via ds_load_b128 WMMA fragments). One wave = 16 query rows.

__global__ void __launch_bounds__(256)
attn_kernel(const u16w* __restrict__ qb, const u16w* __restrict__ kb,
            const u16w* __restrict__ vT, u16w* __restrict__ obuf) {
    __shared__ __align__(16) u16w kstage[2][32 * 32];   // [key row][d]   2 KB each
    __shared__ __align__(16) u16w vstage[2][32 * 32];   // [d row][key]   2 KB each
    __shared__ __align__(16) u16w pstage[8][16 * 32];   // per-wave P staging

    const int lane = threadIdx.x & 31;
    const int wib  = threadIdx.x >> 5;
    const int wg   = blockIdx.x * 8 + wib;
    const int QT   = N_ / 16;              // 128 query tiles per head
    const int bh   = wg / QT;              // uniform across the block (128 % 8 == 0)
    const int qt   = wg % QT;
    const int b = bh >> 3, h = bh & 7;

    const u16w* qbase = qb + ((size_t)bh * N_ + qt * 16) * D_;
    const u16w* kbase = kb + (size_t)bh * N_ * D_;
    const u16w* vbase = vT + (size_t)bh * D_ * N_;
    u16w* pst = &pstage[wib][0];

    v16bf qf = load_frag(qbase, D_, lane);
    v8f zero = {};
    v8f acc0 = zero, acc1 = zero;
    float m_i[8], l_i[8];
#pragma unroll
    for (int r = 0; r < 8; ++r) { m_i[r] = -3.0e38f; l_i[r] = 0.f; }

    const float scale = 0.17677669529663689f;   // 1/sqrt(32)
    const int hi   = lane >> 4;
    const int ncol = lane & 15;

    // issue helpers: K tile is 2048 contiguous bytes (32 rows x 64B);
    // V tile is 32 rows (d) of 64B strided by N*2 bytes.
    auto issueK = [&](int k0, int buf) {
#pragma unroll
        for (int i = 0; i < 4; ++i) {
            u32w chunk = (u32w)(lane + 32 * i);
            async_copy16(&kstage[buf][0], chunk * 16,
                         kbase, (u32w)k0 * 64 + chunk * 16);
        }
    };
    auto issueV = [&](int k0, int buf) {
#pragma unroll
        for (int i = 0; i < 4; ++i) {
            u32w chunk = (u32w)(lane + 32 * i);
            u32w drow = chunk >> 2, jc = chunk & 3;
            async_copy16(&vstage[buf][0], chunk * 16,
                         vbase, drow * (N_ * 2) + (u32w)k0 * 2 + jc * 16);
        }
    };

    int buf = 0;
    if (wib == 0)      issueK(0, buf);
    else if (wib == 1) issueV(0, buf);

    for (int k0 = 0; k0 < N_; k0 += 32) {
        if (wib < 2) wait_asynccnt0();
        __syncthreads();                    // buf ready; buf^1 free for refill
        const int nxt = buf ^ 1;
        if (k0 + 32 < N_) {
            if (wib == 0)      issueK(k0 + 32, nxt);
            else if (wib == 1) issueV(k0 + 32, nxt);
        }

        v16bf kf0 = load_frag(&kstage[buf][0],       32, lane);  // keys k0..k0+15
        v16bf kf1 = load_frag(&kstage[buf][16 * 32], 32, lane);  // keys k0+16..+31
        v8f s0 = wmma_bf16(qf, kf0, zero);
        v8f s1 = wmma_bf16(qf, kf1, zero);
#pragma unroll
        for (int r = 0; r < 8; ++r) {
            float a = s0[r] * scale, c = s1[r] * scale;
            float t = fmaxf(a, c);
            t = fmaxf(t, __shfl_xor(t, 1, 32));
            t = fmaxf(t, __shfl_xor(t, 2, 32));
            t = fmaxf(t, __shfl_xor(t, 4, 32));
            t = fmaxf(t, __shfl_xor(t, 8, 32));
            float mnew = fmaxf(m_i[r], t);
            float corr = __expf(m_i[r] - mnew);
            float p0 = __expf(a - mnew);
            float p1 = __expf(c - mnew);
            float rs = p0 + p1;
            rs += __shfl_xor(rs, 1, 32);
            rs += __shfl_xor(rs, 2, 32);
            rs += __shfl_xor(rs, 4, 32);
            rs += __shfl_xor(rs, 8, 32);
            l_i[r] = l_i[r] * corr + rs;
            m_i[r] = mnew;
            acc0[r] *= corr;
            acc1[r] *= corr;
            int mrow = r + (hi << 3);
            pst[mrow * 32 + ncol]      = f2bf(p0);
            pst[mrow * 32 + 16 + ncol] = f2bf(p1);
        }
        __builtin_amdgcn_wave_barrier();
        asm volatile("s_wait_dscnt 0x0" ::: "memory");
        v16bf pf  = load_frag(pst, 32, lane);                    // A: 16x32 P
        v16bf vf0 = load_frag(&vstage[buf][0],       32, lane);  // B: keys x d0..15
        v16bf vf1 = load_frag(&vstage[buf][16 * 32], 32, lane);  // B: keys x d16..31
        acc0 = wmma_bf16(pf, vf0, acc0);
        acc1 = wmma_bf16(pf, vf1, acc1);
        __builtin_amdgcn_wave_barrier();
        buf = nxt;
    }

    // normalize and store as [B][N][C] bf16 (c = h*32 + d), ready for proj GEMM
#pragma unroll
    for (int r = 0; r < 8; ++r) {
        float rl = 1.f / l_i[r];
        int n = qt * 16 + r + (hi << 3);
        size_t base = ((size_t)b * N_ + n) * C_ + h * D_;
        obuf[base + ncol]      = f2bf(acc0[r] * rl);
        obuf[base + 16 + ncol] = f2bf(acc1[r] * rl);
    }
}

// ---------- kernel 4: projection GEMM + bias + residual ----------

__global__ void __launch_bounds__(256)
proj_gemm_kernel(const u16w* __restrict__ wp, const float* __restrict__ bp,
                 const u16w* __restrict__ obuf, const float* __restrict__ x,
                 float* __restrict__ rbuf) {
    const int lane = threadIdx.x & 31;
    const int wg   = blockIdx.x * 8 + (threadIdx.x >> 5);
    const int b    = wg / (16 * 32);
    const int rem  = wg % (16 * 32);
    const int ot   = rem >> 5;
    const int ng   = rem & 31;
    const int obase = ot << 4;
    const int nbase = ng << 6;

    v8f zero = {};
    v8f acc[4] = {zero, zero, zero, zero};

    const u16w* abase = wp + obase * C_;
    for (int kc = 0; kc < C_; kc += 32) {
        if (kc + 32 < C_) __builtin_prefetch(abase + kc + 32, 0, 3);
        v16bf a = load_frag(abase + kc, C_, lane);
#pragma unroll
        for (int j = 0; j < 4; ++j) {
            const u16w* bb = obuf + (size_t)(b * N_ + nbase + (j << 4)) * C_ + kc;
            v16bf bf = load_frag(bb, C_, lane);
            acc[j] = wmma_bf16(a, bf, acc[j]);
        }
    }

    const int hi   = lane >> 4;
    const int ncol = lane & 15;
#pragma unroll
    for (int r = 0; r < 8; ++r) {
        int o = obase + r + (hi << 3);
        float bias = bp[o];
#pragma unroll
        for (int j = 0; j < 4; ++j) {
            int n = nbase + (j << 4) + ncol;
            size_t idx = ((size_t)b * C_ + o) * N_ + n;
            rbuf[idx] = acc[j][r] + bias + x[idx];
        }
    }
}

// ---------- kernel 5: LayerNorm over channels ----------

__global__ void __launch_bounds__(256)
ln_kernel(const float* __restrict__ rbuf, const float* __restrict__ g,
          const float* __restrict__ be, float* __restrict__ out) {
    __shared__ float red[256];
    const int col = blockIdx.x;            // over B*N columns
    const int b = col / N_, n = col % N_;
    const int c = threadIdx.x;
    const size_t idx = ((size_t)b * C_ + c) * N_ + n;
    float v = rbuf[idx];

    red[c] = v;
    __syncthreads();
#pragma unroll
    for (int s = 128; s > 0; s >>= 1) {
        if (c < s) red[c] += red[c + s];
        __syncthreads();
    }
    float mu = red[0] * (1.0f / C_);
    __syncthreads();

    float dv = v - mu;
    red[c] = dv * dv;
    __syncthreads();
#pragma unroll
    for (int s = 128; s > 0; s >>= 1) {
        if (c < s) red[c] += red[c + s];
        __syncthreads();
    }
    float var = red[0] * (1.0f / C_);
    out[idx] = dv * rsqrtf(var + EPS_) * g[c] + be[c];
}

// ---------- launch ----------

extern "C" void kernel_launch(void* const* d_in, const int* in_sizes, int n_in,
                              void* d_out, int out_size, void* d_ws, size_t ws_size,
                              hipStream_t stream) {
    const float* x     = (const float*)d_in[0];
    const float* wqkv  = (const float*)d_in[1];
    const float* bqkv  = (const float*)d_in[2];
    const float* wproj = (const float*)d_in[3];
    const float* bproj = (const float*)d_in[4];
    const float* lng   = (const float*)d_in[5];
    const float* lnb   = (const float*)d_in[6];
    float* out = (float*)d_out;

    char* ws = (char*)d_ws;
    size_t off = 0;
    auto carve = [&](size_t bytes) {
        char* p = ws + off;
        off += (bytes + 255) & ~(size_t)255;
        return (void*)p;
    };
    u16w* xT   = (u16w*)carve((size_t)B_ * N_ * C_ * 2);        // 4 MB
    u16w* wq16 = (u16w*)carve((size_t)3 * C_ * C_ * 2);         // 384 KB
    u16w* wp16 = (u16w*)carve((size_t)C_ * C_ * 2);             // 128 KB
    u16w* qb   = (u16w*)carve((size_t)B_ * H_ * N_ * D_ * 2);   // 4 MB
    u16w* kbuf = (u16w*)carve((size_t)B_ * H_ * N_ * D_ * 2);   // 4 MB
    u16w* vT   = (u16w*)carve((size_t)B_ * H_ * D_ * N_ * 2);   // 4 MB
    u16w* obuf = (u16w*)carve((size_t)B_ * N_ * C_ * 2);        // 4 MB
    float* rbuf = (float*)carve((size_t)B_ * C_ * N_ * 4);      // 8 MB

    // 1) convert + transpose
    prep_kernel<<<(B_ * C_ * N_) / 256, 256, 0, stream>>>(x, wqkv, wproj, xT, wq16, wp16);
    // 2) QKV GEMM: 4 * 48 * 32 = 6144 waves -> 768 blocks of 8 waves
    qkv_gemm_kernel<<<768, 256, 0, stream>>>(wq16, bqkv, xT, qb, kbuf, vT);
    // 3) attention: 32 bh * 128 qtiles = 4096 waves -> 512 blocks
    attn_kernel<<<512, 256, 0, stream>>>(qb, kbuf, vT, obuf);
    // 4) projection GEMM: 4 * 16 * 32 = 2048 waves -> 256 blocks
    proj_gemm_kernel<<<256, 256, 0, stream>>>(wp16, bproj, obuf, x, rbuf);
    // 5) LayerNorm: one block per (b, n) column
    ln_kernel<<<B_ * N_, 256, 0, stream>>>(rbuf, lng, lnb, out);
}
